// Attention_16947940950099
// MI455X (gfx1250) — compile-verified, matrix operand order checked by reading, not run on hardware
//
#include <hip/hip_runtime.h>
#include <hip/hip_bf16.h>

typedef __attribute__((ext_vector_type(16))) _Float16 v16h;
typedef __attribute__((ext_vector_type(8)))  float    v8f;
typedef __attribute__((ext_vector_type(4)))  unsigned int v4u;
typedef __attribute__((ext_vector_type(8)))  int      v8i;
typedef __attribute__((ext_vector_type(4)))  int      v4i;

#define DIM    1024
#define NHEADS 16
#define HDIM   64
#define BATCH  4
#define SEQ    2048
#define MTOT   (BATCH * SEQ)   /* 8192 */
#define ATTN_SCALE 0.125f      /* 64^-0.5 */

// ---------------------------------------------------------------------------
// TDM: contiguous 1-row tile load (nelem f16 elements) global -> LDS.
// D# built per CDNA5 ISA:
//   group0: [1:0]=count=1, [63:32]=lds_addr, [120:64]=global_addr, [127:126]=2
//   group1: data_size=1 (2B); tensor_dim0=tile_dim0=nelem; dims1..=1;
//           tensor_dim0_stride=nelem
// ---------------------------------------------------------------------------
__device__ __forceinline__ void tdm_load_row_f16(unsigned lds_addr,
                                                 unsigned long long gaddr,
                                                 unsigned nelem) {
  v4u g0;
  g0.x = 1u;                                            // count=1, user mode
  g0.y = lds_addr;                                      // lds_addr
  g0.z = (unsigned)(gaddr & 0xffffffffu);               // global_addr[31:0]
  g0.w = (unsigned)((gaddr >> 32) & 0x01ffffffu)        // global_addr[56:32]
         | 0x80000000u;                                 // type=2 ("image")
  v8i g1;
  g1[0] = 0x00010000;                 // workgroup_mask=0, data_size=2B
  g1[1] = (int)((nelem & 0xffffu) << 16);   // tensor_dim0[15:0] @ bits 48..63
  g1[2] = (int)(((nelem >> 16) & 0xffffu)   // tensor_dim0[31:16] @ bits 64..79
                | (1u << 16));              // tensor_dim1=1 @ bits 80..95
  g1[3] = (int)((nelem & 0xffffu) << 16);   // tile_dim0 @ bits 112..127
  g1[4] = 1;                                // tile_dim1=1
  g1[5] = (int)nelem;                       // tensor_dim0_stride
  g1[6] = 0;
  g1[7] = 0;
#if __has_include(<hip/amd_detail/amd_gfx1250_TDM.h>)
  __builtin_amdgcn_tensor_load_to_lds(g0, g1, (v4i)0, (v4i)0, (v8i)0, 0);
#else
  __builtin_amdgcn_tensor_load_to_lds(g0, g1, (v4i)0, (v4i)0, 0);
#endif
}

// ---------------------------------------------------------------------------
// fp32 -> f16 cast (grid-stride)
// ---------------------------------------------------------------------------
__global__ void mha_cvt_f32_f16(const float* __restrict__ s,
                                _Float16* __restrict__ d, int n) {
  int i = blockIdx.x * blockDim.x + threadIdx.x;
  int stride = gridDim.x * blockDim.x;
  for (; i < n; i += stride) d[i] = (_Float16)s[i];
}

// ---------------------------------------------------------------------------
// C[m,n] = sum_k A[m,k] * W[n,k] + bias[n]   (one 16x16 tile per wave)
// mode 0: f16 out in [B,H,S,Dh]; mode 1: f32 out in [MTOT,DIM]
// ---------------------------------------------------------------------------
__global__ void mha_gemm_wmma(const _Float16* __restrict__ A,
                              const _Float16* __restrict__ W,
                              const float* __restrict__ bias,
                              void* __restrict__ out, int mode) {
  const int lane  = threadIdx.x & 31;
  const int wave  = threadIdx.x >> 5;
  const int tile  = blockIdx.x * 8 + wave;
  const int tile_m = tile >> 6;
  const int tile_n = tile & 63;

  const int row = tile_m * 16 + (lane & 15);
  const int col = tile_n * 16 + (lane & 15);
  const int kh  = (lane >> 4) * 16;

  const _Float16* aptr = A + (size_t)row * DIM + kh;
  const _Float16* wptr = W + (size_t)col * DIM + kh;

  v8f acc = {};
  for (int k = 0; k < DIM; k += 32) {
    __builtin_prefetch(aptr + 8 * 32, 0, 1);
    __builtin_prefetch(wptr + 8 * 32, 0, 1);
    v16h a = *(const v16h*)aptr;
    v16h b = *(const v16h*)wptr;
    acc = __builtin_amdgcn_wmma_f32_16x16x32_f16(
        false, a, false, b, (short)0, acc, false, false);
    aptr += 32;
    wptr += 32;
  }

  const float bn = bias[col];
  const int mhi = (lane >> 4) << 3;

  if (mode == 0) {
    _Float16* O = (_Float16*)out;
    const int h  = col >> 6;
    const int dh = col & 63;
#pragma unroll
    for (int r = 0; r < 8; ++r) {
      int m = tile_m * 16 + r + mhi;
      int b = m >> 11;
      int s = m & 2047;
      O[(((size_t)b * NHEADS + h) * SEQ + s) * HDIM + dh] =
          (_Float16)(acc[r] + bn);
    }
  } else {
    float* O = (float*)out;
#pragma unroll
    for (int r = 0; r < 8; ++r) {
      int m = tile_m * 16 + r + mhi;
      O[(size_t)m * DIM + col] = acc[r] + bn;
    }
  }
}

// ---------------------------------------------------------------------------
// Flash attention for one (b,h); 8 waves x 16 query rows per block.
// K tile staged via TDM tensor_load_to_lds (TENSORcnt);
// V tile staged via global_load_async_to_lds_b128 (ASYNCcnt), then
// transposed in LDS; S=QK^T and O+=P@V via v_wmma_f32_16x16x32_f16.
// ---------------------------------------------------------------------------
__global__ void mha_attn_wmma(const _Float16* __restrict__ Q,
                              const _Float16* __restrict__ K,
                              const _Float16* __restrict__ V,
                              _Float16* __restrict__ O) {
  __shared__ _Float16 kT[32 * 64];      // K tile   [key][dh]  (TDM dest)
  __shared__ _Float16 vR[32 * 64];      // V tile   [key][dh]  (async dest)
  __shared__ _Float16 vT[64 * 32];      // V^T tile [dh][key]
  __shared__ _Float16 pb[8][16 * 32];   // per-wave P staging

  const int tid  = threadIdx.x;
  const int lane = tid & 31;
  const int wave = tid >> 5;
  const int bh   = blockIdx.x >> 4;
  const int qblk = blockIdx.x & 15;
  const int q0   = qblk * 128 + wave * 16;
  const int b    = bh >> 4;
  const int h    = bh & 15;
  const size_t base = (size_t)bh * SEQ * HDIM;

  const int arow = lane & 15;
  const int kh   = (lane >> 4) * 16;
  const int mhi  = (lane >> 4) << 3;

  const unsigned kT_lds = (unsigned)(uintptr_t)&kT[0];
  const unsigned vR_lds = (unsigned)(uintptr_t)&vR[0];

  v16h qa0 = *(const v16h*)(Q + base + (size_t)(q0 + arow) * HDIM + kh);
  v16h qa1 = *(const v16h*)(Q + base + (size_t)(q0 + arow) * HDIM + 32 + kh);

  v8f o0 = {}, o1 = {}, o2 = {}, o3 = {};
  float rmax[8], rsum[8];
#pragma unroll
  for (int r = 0; r < 8; ++r) { rmax[r] = -1e30f; rsum[r] = 0.0f; }

  for (int j = 0; j < SEQ; j += 32) {
    __syncthreads();                      // previous-iter consumers done

    // --- K tile: one TDM descriptor covers the contiguous 32x64 tile ---
    if (wave == 0) {
      tdm_load_row_f16(kT_lds,
                       (unsigned long long)(uintptr_t)(K + base +
                                                       (size_t)j * HDIM),
                       32u * 64u);
      __builtin_amdgcn_s_wait_tensorcnt(0);
    }
    // --- V tile: async global->LDS, 16B per thread (ASYNCcnt) ---
    {
      unsigned lds_dst = vR_lds + (unsigned)tid * 16u;
      unsigned long long gsrc =
          (unsigned long long)(uintptr_t)(V + base + (size_t)j * HDIM) +
          (unsigned long long)tid * 16ull;
      asm volatile("global_load_async_to_lds_b128 %0, %1, off"
                   :: "v"(lds_dst), "v"(gsrc) : "memory");
      asm volatile("s_wait_asynccnt 0x0" ::: "memory");
    }
    __syncthreads();                      // K & V tiles resident in LDS

    // --- transpose V in LDS: vR[key][dh] -> vT[dh][key] ---
    {
      int vrow = tid >> 3;                // key 0..31
      int vcol = (tid & 7) * 8;           // dh start
      _Float16 tmp[8];
      *(float4*)tmp = *(const float4*)(vR + vrow * HDIM + vcol);
#pragma unroll
      for (int i = 0; i < 8; ++i) vT[(vcol + i) * 32 + vrow] = tmp[i];
    }
    __syncthreads();

    // --- S = Q @ K^T for 32 keys ---
    const _Float16* kp0 = kT + (size_t)(lane & 15) * HDIM + kh;
    const _Float16* kp1 = kT + (size_t)(16 + (lane & 15)) * HDIM + kh;
    v8f s0 = {}, s1 = {};
    s0 = __builtin_amdgcn_wmma_f32_16x16x32_f16(
        false, qa0, false, *(const v16h*)kp0, (short)0, s0, false, false);
    s0 = __builtin_amdgcn_wmma_f32_16x16x32_f16(
        false, qa1, false, *(const v16h*)(kp0 + 32), (short)0, s0, false, false);
    s1 = __builtin_amdgcn_wmma_f32_16x16x32_f16(
        false, qa0, false, *(const v16h*)kp1, (short)0, s1, false, false);
    s1 = __builtin_amdgcn_wmma_f32_16x16x32_f16(
        false, qa1, false, *(const v16h*)(kp1 + 32), (short)0, s1, false, false);

    // --- online softmax (16-lane row groups) ---
#pragma unroll
    for (int r = 0; r < 8; ++r) {
      float a0 = s0[r] * ATTN_SCALE;
      float a1 = s1[r] * ATTN_SCALE;
      float mx = fmaxf(a0, a1);
#pragma unroll
      for (int off = 1; off < 16; off <<= 1)
        mx = fmaxf(mx, __shfl_xor(mx, off, 32));
      float mnew  = fmaxf(rmax[r], mx);
      float alpha = __expf(rmax[r] - mnew);
      rmax[r] = mnew;
      float p0 = __expf(a0 - mnew);
      float p1 = __expf(a1 - mnew);
      float ps = p0 + p1;
#pragma unroll
      for (int off = 1; off < 16; off <<= 1)
        ps += __shfl_xor(ps, off, 32);
      rsum[r] = rsum[r] * alpha + ps;
      o0[r] *= alpha; o1[r] *= alpha; o2[r] *= alpha; o3[r] *= alpha;
      int m = r + mhi;
      pb[wave][m * 32 + (lane & 15)]      = (_Float16)p0;
      pb[wave][m * 32 + 16 + (lane & 15)] = (_Float16)p1;
    }
    asm volatile("s_wait_dscnt 0" ::: "memory");

    // --- O += P (16x32) @ V (32x64) ---
    v16h pa = *(const v16h*)(&pb[wave][(size_t)(lane & 15) * 32 + kh]);
    o0 = __builtin_amdgcn_wmma_f32_16x16x32_f16(
        false, pa, false,
        *(const v16h*)(vT + (0 * 16 + (lane & 15)) * 32 + kh),
        (short)0, o0, false, false);
    o1 = __builtin_amdgcn_wmma_f32_16x16x32_f16(
        false, pa, false,
        *(const v16h*)(vT + (1 * 16 + (lane & 15)) * 32 + kh),
        (short)0, o1, false, false);
    o2 = __builtin_amdgcn_wmma_f32_16x16x32_f16(
        false, pa, false,
        *(const v16h*)(vT + (2 * 16 + (lane & 15)) * 32 + kh),
        (short)0, o2, false, false);
    o3 = __builtin_amdgcn_wmma_f32_16x16x32_f16(
        false, pa, false,
        *(const v16h*)(vT + (3 * 16 + (lane & 15)) * 32 + kh),
        (short)0, o3, false, false);
    asm volatile("s_wait_dscnt 0" ::: "memory");
  }

  // --- normalize and store O in [B,S,D] f16 layout ---
  const int col = lane & 15;
#pragma unroll
  for (int r = 0; r < 8; ++r) {
    float inv = 1.0f / rsum[r];
    int m = q0 + r + mhi;
    size_t orow = ((size_t)b * SEQ + m) * DIM + (size_t)h * HDIM;
    O[orow + 0 * 16 + col] = (_Float16)(o0[r] * inv);
    O[orow + 1 * 16 + col] = (_Float16)(o1[r] * inv);
    O[orow + 2 * 16 + col] = (_Float16)(o2[r] * inv);
    O[orow + 3 * 16 + col] = (_Float16)(o3[r] * inv);
  }
}

// ---------------------------------------------------------------------------
extern "C" void kernel_launch(void* const* d_in, const int* in_sizes, int n_in,
                              void* d_out, int out_size, void* d_ws,
                              size_t ws_size, hipStream_t stream) {
  (void)in_sizes; (void)n_in; (void)out_size; (void)ws_size;
  const float* x  = (const float*)d_in[0];
  const float* Wq = (const float*)d_in[1];
  const float* bq = (const float*)d_in[2];
  const float* Wk = (const float*)d_in[3];
  const float* bk = (const float*)d_in[4];
  const float* Wv = (const float*)d_in[5];
  const float* bv = (const float*)d_in[6];
  const float* Wo = (const float*)d_in[7];
  const float* bo = (const float*)d_in[8];

  _Float16* xh  = (_Float16*)d_ws;
  _Float16* wqh = xh  + (size_t)MTOT * DIM;
  _Float16* wkh = wqh + (size_t)DIM * DIM;
  _Float16* wvh = wkh + (size_t)DIM * DIM;
  _Float16* woh = wvh + (size_t)DIM * DIM;
  _Float16* Qh  = woh + (size_t)DIM * DIM;
  _Float16* Kh  = Qh  + (size_t)MTOT * DIM;
  _Float16* Vh  = Kh  + (size_t)MTOT * DIM;
  _Float16* Oh  = Vh  + (size_t)MTOT * DIM;

  mha_cvt_f32_f16<<<1024, 256, 0, stream>>>(x,  xh,  MTOT * DIM);
  mha_cvt_f32_f16<<<256,  256, 0, stream>>>(Wq, wqh, DIM * DIM);
  mha_cvt_f32_f16<<<256,  256, 0, stream>>>(Wk, wkh, DIM * DIM);
  mha_cvt_f32_f16<<<256,  256, 0, stream>>>(Wv, wvh, DIM * DIM);
  mha_cvt_f32_f16<<<256,  256, 0, stream>>>(Wo, woh, DIM * DIM);

  mha_gemm_wmma<<<4096, 256, 0, stream>>>(xh, wqh, bq, (void*)Qh, 0);
  mha_gemm_wmma<<<4096, 256, 0, stream>>>(xh, wkh, bk, (void*)Kh, 0);
  mha_gemm_wmma<<<4096, 256, 0, stream>>>(xh, wvh, bv, (void*)Vh, 0);

  mha_attn_wmma<<<1024, 256, 0, stream>>>(Qh, Kh, Vh, Oh);

  mha_gemm_wmma<<<4096, 256, 0, stream>>>(Oh, woh, bo, d_out, 1);
}
